// IGAVNet72_71116068487802
// MI455X (gfx1250) — matrix operand order repaired and checked by reading, not compiled, not run on hardware
//
#include <hip/hip_runtime.h>
#include <hip/hip_bf16.h>

// ---------------------------------------------------------------------------
// Types for CDNA5 WMMA (wave32)
// ---------------------------------------------------------------------------
typedef __attribute__((ext_vector_type(16))) _Float16 v16h;
typedef __attribute__((ext_vector_type(8)))  _Float16 v8h;
typedef __attribute__((ext_vector_type(8)))  float    v8f;
typedef int v4i __attribute__((vector_size(16)));   // matches builtin param type

#define DEVI __device__ __forceinline__
#define AS1 __attribute__((address_space(1)))
#define AS3 __attribute__((address_space(3)))

static constexpr int HW   = 108;
static constexpr int NPIX = HW * HW;        // 11664
static constexpr int C    = 64;
static constexpr int NB   = 9;              // blocks (3x3 of 36x36)
static constexpr int NN   = 36 * 36;        // 1296 nodes per block
static constexpr int NNP  = 1312;           // NN padded to K-chunk multiple (32)

// ---------------------------------------------------------------------------
// Async global->LDS staging (CDNA5 GLOBAL_LOAD_ASYNC_TO_LDS_B128, ASYNCcnt)
// with compile-safe fallback to plain 16B copies.
// ---------------------------------------------------------------------------
#if defined(__HIP_DEVICE_COMPILE__) && \
    __has_builtin(__builtin_amdgcn_global_load_async_to_lds_b128) && \
    __has_builtin(__builtin_amdgcn_s_wait_asynccnt)
#define HAVE_ASYNC 1
#else
#define HAVE_ASYNC 0
#endif
#if defined(__HIP_DEVICE_COMPILE__) && !HAVE_ASYNC
#warning "gfx1250 async-to-LDS builtins not found; falling back to VALU staging"
#endif

DEVI void cp16(void* lds, const void* g) {
#if HAVE_ASYNC
    __builtin_amdgcn_global_load_async_to_lds_b128(
        (AS1 v4i*)g, (AS3 v4i*)lds, 0, 0);
#else
    *(v8h*)lds = *(const v8h*)g;   // 16-byte copy through VGPRs
#endif
}

DEVI void async_wait() {
#if HAVE_ASYNC
    __builtin_amdgcn_s_wait_asynccnt(0);
#endif
}

// ---------------------------------------------------------------------------
// WMMA helpers: 16x16x32 f16 -> f32
// A tile in LDS: [16 rows(M)][32 k]; lane (m=l&15, kh=l>>4) holds
// K = {kh*8..+7} U {16+kh*8..+7}  (ISA 7.12.2 16-bit A layout).
// B tile in LDS stored transposed: Bt[16 cols(N)][32 k]; lane (n=l&15,
// kh=l>>4) holds K = {kh*8..+7} U {16+kh*8..+7} (same order as A).
// ---------------------------------------------------------------------------
DEVI v16h load_frag_a(const _Float16 (&T)[16][32], int lane) {
    int m = lane & 15, kh = (lane >> 4) & 1;
    const _Float16* p = &T[m][kh * 8];
    v8h lo = *(const v8h*)(p);
    v8h hi = *(const v8h*)(p + 16);
    v16h r;
#pragma unroll
    for (int i = 0; i < 8; ++i) { r[i] = lo[i]; r[i + 8] = hi[i]; }
    return r;
}

DEVI v16h load_frag_b(const _Float16 (&T)[16][32], int lane) {
    int n = lane & 15, kh = (lane >> 4) & 1;
    const _Float16* p = &T[n][kh * 8];
    v8h lo = *(const v8h*)(p);
    v8h hi = *(const v8h*)(p + 16);
    v16h r;
#pragma unroll
    for (int i = 0; i < 8; ++i) { r[i] = lo[i]; r[i + 8] = hi[i]; }
    return r;
}

DEVI v8f wmma_f16(v16h a, v16h b, v8f c) {
    return __builtin_amdgcn_wmma_f32_16x16x32_f16(
        false, a, false, b, (short)0, c, false, false);
}

// ---------------------------------------------------------------------------
// Staging / reshape kernels
// ---------------------------------------------------------------------------
__global__ void k_f32_to_f16(const float* __restrict__ in, _Float16* __restrict__ out, int n) {
    int i = blockIdx.x * 256 + threadIdx.x;
    if (i < n) out[i] = (_Float16)in[i];
}

// w (K x N, f32) -> wT (N x K, f16)
__global__ void k_wT(const float* __restrict__ w, _Float16* __restrict__ wT, int K, int N) {
    int i = blockIdx.x * 256 + threadIdx.x;
    if (i >= N * K) return;
    int k = i % K, n = i / K;
    wT[i] = (_Float16)w[(size_t)k * N + n];
}

// z [b*n][H*Fo] -> zT [(b*H+h)*Fo+f][NNP], zero pad n in [NN,NNP)
__global__ void k_zT(const _Float16* __restrict__ z, _Float16* __restrict__ zT, int H, int Fo) {
    int i = blockIdx.x * 256 + threadIdx.x;
    int total = NB * H * Fo * NNP;
    if (i >= total) return;
    int n = i % NNP, row = i / NNP;
    int f = row % Fo, bh = row / Fo;
    int h = bh % H, b = bh / H;
    _Float16 v = (_Float16)0.f;
    if (n < NN) v = z[((size_t)(b * NN + n) * H + h) * Fo + f];
    zT[i] = v;
}

// ---------------------------------------------------------------------------
// 3x3 conv (pad=1) as implicit GEMM. K = 576 (ci*9+ky*3+kx).
// One wave -> 16 cout x 64 positions (4 WMMA tiles, shared A fragment).
// ---------------------------------------------------------------------------
__global__ void k_conv3x3(const _Float16* __restrict__ in,   // [64][NPIX] f16
                          const _Float16* __restrict__ w,    // [64][576] f16 (OIHW flat)
                          const float*    __restrict__ bias, // [64]
                          float*          __restrict__ out)  // [64][NPIX]
{
    __shared__ _Float16 At[16][32];
    __shared__ _Float16 Bt[4][16][32];
    const int lane = threadIdx.x;
    const int pb = blockIdx.x * 64;  // 183 blocks of 4 pos tiles
    const int c0 = blockIdx.y * 16;  // 4 cout tiles
    v8f acc[4] = {};
    for (int k0 = 0; k0 < 576; k0 += 32) {
        // A: weights, contiguous rows -> async b128
#pragma unroll
        for (int j = 0; j < 2; ++j) {
            int e = j * 32 + lane, r = e >> 2, s = e & 3;
            cp16(&At[r][s * 8], w + (size_t)(c0 + r) * 576 + k0 + s * 8);
        }
        // B: on-the-fly im2col gather (4 tiles)
        for (int idx = lane; idx < 2048; idx += 32) {
            int t = idx >> 9, rem = idx & 511;
            int n = rem >> 5, k = rem & 31;
            int kk = k0 + k;
            int ci = kk / 9, tt = kk - ci * 9;
            int ky = tt / 3, kx = tt - ky * 3;
            int p = pb + t * 16 + n;
            _Float16 v = (_Float16)0.f;
            if (p < NPIX) {
                int y = p / HW, x = p - y * HW;
                int iy = y + ky - 1, ix = x + kx - 1;
                if ((unsigned)iy < (unsigned)HW && (unsigned)ix < (unsigned)HW)
                    v = in[ci * NPIX + iy * HW + ix];
            }
            Bt[t][n][k] = v;
        }
        async_wait();
        __syncthreads();
        v16h a = load_frag_a(At, lane);
#pragma unroll
        for (int t = 0; t < 4; ++t)
            acc[t] = wmma_f16(a, load_frag_b(Bt[t], lane), acc[t]);
        __syncthreads();
    }
    int n = lane & 15, mh = lane >> 4;
#pragma unroll
    for (int t = 0; t < 4; ++t) {
#pragma unroll
        for (int e = 0; e < 8; ++e) {
            int co = c0 + e + 8 * mh;
            int p = pb + t * 16 + n;
            if (p < NPIX) out[co * NPIX + p] = acc[t][e] + bias[co];
        }
    }
}

// ---------------------------------------------------------------------------
// Generic batched GEMM: Z[b] = A[b] (MxK) * B (KxN) with B pre-transposed
// (BwT is N x K). One wave -> 16 x 64 output (4 WMMA tiles).
// M,K mult of 16/32; N mult of 64. Fully async-staged operands.
// ---------------------------------------------------------------------------
__global__ void k_gemm_f16(const _Float16* __restrict__ A, const _Float16* __restrict__ BwT,
                           _Float16* __restrict__ Z, int M, int N, int K) {
    __shared__ _Float16 At[16][32];
    __shared__ _Float16 Bt[4][16][32];
    int lane = threadIdx.x;
    int m0 = blockIdx.x * 16, n0 = blockIdx.y * 64, b = blockIdx.z;
    const _Float16* Ab = A + (size_t)b * M * K;
    v8f acc[4] = {};
    for (int k0 = 0; k0 < K; k0 += 32) {
#pragma unroll
        for (int j = 0; j < 2; ++j) {
            int e = j * 32 + lane, r = e >> 2, s = e & 3;
            cp16(&At[r][s * 8], Ab + (size_t)(m0 + r) * K + k0 + s * 8);
        }
#pragma unroll
        for (int t = 0; t < 4; ++t)
#pragma unroll
            for (int j = 0; j < 2; ++j) {
                int e = j * 32 + lane, r = e >> 2, s = e & 3;
                cp16(&Bt[t][r][s * 8], BwT + (size_t)(n0 + t * 16 + r) * K + k0 + s * 8);
            }
        if (k0 + 32 < K)  // speculative prefetch of next A chunk
            __builtin_prefetch(Ab + (size_t)(m0 + (lane & 15)) * K + k0 + 32);
        async_wait();
        __syncthreads();
        v16h a = load_frag_a(At, lane);
#pragma unroll
        for (int t = 0; t < 4; ++t)
            acc[t] = wmma_f16(a, load_frag_b(Bt[t], lane), acc[t]);
        __syncthreads();
    }
    int n = lane & 15, mh = lane >> 4;
#pragma unroll
    for (int t = 0; t < 4; ++t)
#pragma unroll
        for (int e = 0; e < 8; ++e) {
            int m = m0 + e + 8 * mh;
            Z[(size_t)b * M * N + (size_t)m * N + n0 + t * 16 + n] = (_Float16)acc[t][e];
        }
}

// ---------------------------------------------------------------------------
// Pointwise / reduction kernels
// ---------------------------------------------------------------------------
__global__ void k_bn_stats(const float* __restrict__ x, const float* __restrict__ g,
                           const float* __restrict__ be, float* __restrict__ scale,
                           float* __restrict__ shift) {
    __shared__ float ssum[256], ssq[256];
    int c = blockIdx.x, t = threadIdx.x;
    float s = 0.f, q = 0.f;
    for (int i = t; i < NPIX; i += 256) {
        float v = x[c * NPIX + i];
        s += v; q += v * v;
    }
    ssum[t] = s; ssq[t] = q;
    __syncthreads();
    for (int o = 128; o > 0; o >>= 1) {
        if (t < o) { ssum[t] += ssum[t + o]; ssq[t] += ssq[t + o]; }
        __syncthreads();
    }
    if (t == 0) {
        float m = ssum[0] / (float)NPIX;
        float var = ssq[0] / (float)NPIX - m * m;
        float sc = g[c] * rsqrtf(var + 1e-5f);
        scale[c] = sc;
        shift[c] = be[c] - m * sc;
    }
}

__global__ void k_bn_relu(const float* __restrict__ x, const float* __restrict__ scale,
                          const float* __restrict__ shift, float* __restrict__ of32,
                          _Float16* __restrict__ of16) {
    int i = blockIdx.x * 256 + threadIdx.x;
    if (i >= C * NPIX) return;
    int c = i / NPIX;
    float v = x[i] * scale[c] + shift[c];
    v = v > 0.f ? v : 0.f;
    if (of32) of32[i] = v;
    of16[i] = (_Float16)v;
}

__global__ void k_am(const float* __restrict__ x, const float* __restrict__ w,
                     float* __restrict__ out) {
    int p = blockIdx.x * 256 + threadIdx.x;
    if (p >= NPIX) return;
    float a = 0.f;
    for (int c = 0; c < C; ++c) a += x[c * NPIX + p] * w[c];
    a = a > 0.f ? a : 0.01f * a;
    for (int c = 0; c < C; ++c) out[c * NPIX + p] = x[c * NPIX + p] * a;
}

__global__ void k_pool(const float* __restrict__ x, float* __restrict__ pooled) {
    __shared__ float sm[256], mx[256];
    int c = blockIdx.x, t = threadIdx.x;
    float s = 0.f, m = -1e30f;
    for (int i = t; i < NPIX; i += 256) {
        float v = x[c * NPIX + i];
        s += v; m = fmaxf(m, v);
    }
    sm[t] = s; mx[t] = m;
    __syncthreads();
    for (int o = 128; o > 0; o >>= 1) {
        if (t < o) { sm[t] += sm[t + o]; mx[t] = fmaxf(mx[t], mx[t + o]); }
        __syncthreads();
    }
    if (t == 0) pooled[c] = sm[0] / (float)NPIX + mx[0];
}

__global__ void k_mlp(const float* __restrict__ pooled, const float* __restrict__ w1,
                      const float* __restrict__ b1, const float* __restrict__ w2,
                      const float* __restrict__ b2, float* __restrict__ s) {
    __shared__ float hid[16];
    int t = threadIdx.x;
    if (t < 16) {
        float a = b1[t];
        for (int c = 0; c < 64; ++c) a += w1[t * 64 + c] * pooled[c];
        hid[t] = a > 0.f ? a : 0.f;
    }
    __syncthreads();
    float a = b2[t];
    for (int k = 0; k < 16; ++k) a += w2[t * 16 + k] * hid[k];
    s[t] = 1.f / (1.f + __expf(-a));
}

__global__ void k_scale_f16(const float* __restrict__ x, const float* __restrict__ s,
                            _Float16* __restrict__ out) {
    int i = blockIdx.x * 256 + threadIdx.x;
    if (i >= C * NPIX) return;
    int c = i / NPIX;
    out[i] = (_Float16)(x[i] * s[c]);
}

__global__ void k_blockify(const float* __restrict__ x2, _Float16* __restrict__ feat,
                           float* __restrict__ v) {
    int i = blockIdx.x * 256 + threadIdx.x;
    if (i >= NB * NN) return;
    int b = i / NN, n = i - b * NN;
    int bx = b / 3, by = b - bx * 3;
    int sy = n / 36, sx = n - sy * 36;
    int p = (bx * 36 + sy) * HW + (by * 36 + sx);
    float s = 0.f;
    for (int c = 0; c < C; ++c) {
        float val = x2[c * NPIX + p];
        feat[(size_t)i * C + c] = (_Float16)val;
        s += val;
    }
    v[i] = s * (1.f / 64.f);
}

__global__ void k_eler(const _Float16* __restrict__ z, const float* __restrict__ al,
                       const float* __restrict__ ar, float* __restrict__ el,
                       float* __restrict__ er, int H, int Fo) {
    int i = blockIdx.x * 256 + threadIdx.x;
    int total = NB * NN * H;
    if (i >= total) return;
    int h = i % H, bn = i / H;
    const _Float16* zp = z + ((size_t)bn * H + h) * Fo;
    float sl = 0.f, sr = 0.f;
    for (int f = 0; f < Fo; ++f) {
        float zv = (float)zp[f];
        sl += zv * al[h * Fo + f];
        sr += zv * ar[h * Fo + f];
    }
    el[i] = sl;
    er[i] = sr;
}

// Masked softmax stats over i for each (b,j,h): online max + weighted denom.
__global__ void k_attn_stats(const float* __restrict__ el, const float* __restrict__ er,
                             const float* __restrict__ v, float* __restrict__ mOut,
                             float* __restrict__ dOut, int H) {
    int i = blockIdx.x * 256 + threadIdx.x;
    int total = NB * NN * H;
    if (i >= total) return;
    int h = i % H, bj = i / H;
    int b = bj / NN, j = bj - b * NN;
    float erj = er[i], vj = v[bj];
    float m = -1e30f, s = 0.f;
    const float* elb = el + ((size_t)b * NN) * H + h;
    const float* vb = v + b * NN;
    for (int ii = 0; ii < NN; ++ii) {
        float wgt = (vb[ii] * vj > 0.25f ? 1.f : 0.f) + (ii == j ? 1.f : 0.f);
        if (wgt > 0.f) {
            float e = elb[(size_t)ii * H] + erj;
            e = e > 0.f ? e : 0.2f * e;
            if (e > m) { s = s * __expf(m - e) + wgt; m = e; }
            else       { s += wgt * __expf(e - m); }
        }
    }
    mOut[i] = m;
    dOut[i] = s;
}

// ---------------------------------------------------------------------------
// out[b,j,h,f] = sum_i alpha[i,j] * z[b,i,h,f]  (+bias, +activation)
// alpha tiles generated procedurally in LDS (flash style); z staged async
// from zT ([(b*H+h)*Fo+f][NNP], zero-padded). One wave -> 16 j x 64 f.
// ---------------------------------------------------------------------------
__global__ void k_attn_out(const _Float16* __restrict__ zT, const float* __restrict__ el,
                           const float* __restrict__ er, const float* __restrict__ v,
                           const float* __restrict__ m, const float* __restrict__ den,
                           const float* __restrict__ bias, _Float16* __restrict__ outF16,
                           float* __restrict__ outF32, int H, int Fo, int act) {
    __shared__ _Float16 At[16][32];     // alpha^T: [j][i]
    __shared__ _Float16 Bt[4][16][32];  // z^T:     [f][i]
    __shared__ float s_er[16], s_v[16], s_m[16], s_d[16];
    __shared__ float s_el[32], s_vi[32];
    int lane = threadIdx.x;
    int j0 = blockIdx.x * 16, f0 = blockIdx.y * 64;
    int bh = blockIdx.z;
    int b = bh / H, h = bh - b * H;
    if (lane < 16) {
        int bj = b * NN + j0 + lane;
        s_er[lane] = er[(size_t)bj * H + h];
        s_v[lane]  = v[bj];
        s_m[lane]  = m[(size_t)bj * H + h];
        s_d[lane]  = den[(size_t)bj * H + h];
    }
    __syncthreads();
    const _Float16* zrow = zT + ((size_t)(b * H + h) * Fo + f0) * NNP;
    v8f acc[4] = {};
    for (int i0 = 0; i0 < NN; i0 += 32) {
        {   // per-chunk row stats
            int ii = i0 + lane;
            if (ii < NN) {
                s_el[lane] = el[((size_t)b * NN + ii) * H + h];
                s_vi[lane] = v[b * NN + ii];
            } else { s_el[lane] = 0.f; s_vi[lane] = 0.f; }
        }
        // z tiles: contiguous rows of zT -> async b128
#pragma unroll
        for (int t = 0; t < 4; ++t)
#pragma unroll
            for (int j = 0; j < 2; ++j) {
                int e = j * 32 + lane, r = e >> 2, s = e & 3;
                cp16(&Bt[t][r][s * 8], zrow + (size_t)(t * 16 + r) * NNP + i0 + s * 8);
            }
        __syncthreads();
        // procedural alpha tile
        for (int idx = lane; idx < 512; idx += 32) {
            int j = idx >> 5, i = idx & 31;
            int ii = i0 + i;
            float a = 0.f;
            if (ii < NN) {
                float wgt = (s_vi[i] * s_v[j] > 0.25f ? 1.f : 0.f) + (ii == j0 + j ? 1.f : 0.f);
                if (wgt > 0.f) {
                    float e = s_el[i] + s_er[j];
                    e = e > 0.f ? e : 0.2f * e;
                    a = wgt * __expf(e - s_m[j]) / s_d[j];
                }
            }
            At[j][i] = (_Float16)a;
        }
        async_wait();
        __syncthreads();
        v16h a = load_frag_a(At, lane);
#pragma unroll
        for (int t = 0; t < 4; ++t)
            acc[t] = wmma_f16(a, load_frag_b(Bt[t], lane), acc[t]);
        __syncthreads();
    }
    int fl = lane & 15, mh = lane >> 4;
#pragma unroll
    for (int t = 0; t < 4; ++t)
#pragma unroll
        for (int e = 0; e < 8; ++e) {
            int j = j0 + e + 8 * mh;
            int f = f0 + t * 16 + fl;
            float val = acc[t][e] + bias[h * Fo + f];
            if (act == 1) val = val > 0.f ? val : (__expf(val) - 1.f);  // elu
            size_t o = (size_t)b * NN + j;
            if (outF16) outF16[o * (H * Fo) + h * Fo + f] = (_Float16)val;
            if (outF32) outF32[(o * H + h) * Fo + f] = val;
        }
}

__global__ void k_head_mean(const float* __restrict__ t, float* __restrict__ o) {
    int i = blockIdx.x * 256 + threadIdx.x;
    if (i >= NB * NN * 64) return;
    int c = i & 63, bn = i >> 6;
    o[i] = (t[(size_t)bn * 192 + c] + t[(size_t)bn * 192 + 64 + c] +
            t[(size_t)bn * 192 + 128 + c]) * (1.f / 3.f);
}

__global__ void k_uperm(const float* __restrict__ hbar, _Float16* __restrict__ uh) {
    int i = blockIdx.x * 256 + threadIdx.x;
    if (i >= C * NPIX) return;
    int c = i / NPIX, q = i - c * NPIX;
    uh[i] = (_Float16)hbar[(size_t)q * C + c];
}

__global__ void k_final(const float* __restrict__ x1f, const float* __restrict__ uf,
                        const float* __restrict__ we, float* __restrict__ out) {
    int i = blockIdx.x * 256 + threadIdx.x;
    if (i >= C * NPIX) return;
    int co = i / NPIX, p = i - co * NPIX;
    float a = 0.f;
    for (int ci = 0; ci < C; ++ci) a += x1f[ci * NPIX + p] * we[co * 128 + ci];
    for (int ci = 0; ci < C; ++ci) a += uf[ci * NPIX + p] * we[co * 128 + 64 + ci];
    out[i] = a > 0.f ? a : 0.01f * a;
}

// ---------------------------------------------------------------------------
// Host orchestration
// ---------------------------------------------------------------------------
enum {
    IN_X = 0,
    IN_C1_W1, IN_C1_B1, IN_C1_G1, IN_C1_BE1, IN_C1_W2, IN_C1_B2, IN_C1_G2, IN_C1_BE2,
    IN_AM11_W,
    IN_MLP_W1, IN_MLP_B1, IN_MLP_W2, IN_MLP_B2,
    IN_C12_W1, IN_C12_B1, IN_C12_G1, IN_C12_BE1, IN_C12_W2, IN_C12_B2, IN_C12_G2, IN_C12_BE2,
    IN_AM21_W,
    IN_G1_W, IN_G1_AL, IN_G1_AR, IN_G1_B,
    IN_G2_W, IN_G2_AL, IN_G2_AR, IN_G2_B,
    IN_G3_W, IN_G3_AL, IN_G3_AR, IN_G3_B,
    IN_C22_W1, IN_C22_B1, IN_C22_G1, IN_C22_BE1, IN_C22_W2, IN_C22_B2, IN_C22_G2, IN_C22_BE2,
    IN_AM_END_W
};

struct Bump {
    char* base;
    size_t used;
    template <class T> T* get(size_t n) {
        size_t bytes = (n * sizeof(T) + 255) & ~(size_t)255;
        T* r = (T*)(base + used);
        used += bytes;
        return r;
    }
};

static inline dim3 g1(int n) { return dim3((n + 255) / 256); }

extern "C" void kernel_launch(void* const* d_in, const int* in_sizes, int n_in,
                              void* d_out, int out_size, void* d_ws, size_t ws_size,
                              hipStream_t stream) {
    (void)in_sizes; (void)n_in; (void)out_size; (void)ws_size;
    auto F = [&](int i) { return (const float*)d_in[i]; };

    Bump ws{(char*)d_ws, 0};
    // f16 staging
    _Float16* x0h = ws.get<_Float16>(C * NPIX);
    _Float16* wc[6];
    for (int i = 0; i < 6; ++i) wc[i] = ws.get<_Float16>(36864);
    _Float16* gwT1 = ws.get<_Float16>(384 * 64);
    _Float16* gwT2 = ws.get<_Float16>(640 * 384);
    _Float16* gwT3 = ws.get<_Float16>(192 * 640);
    // fp32 buffers
    float* tconv  = ws.get<float>(C * NPIX);
    float* scale  = ws.get<float>(64);
    float* shift  = ws.get<float>(64);
    float* xf     = ws.get<float>(C * NPIX);
    float* x1     = ws.get<float>(C * NPIX);
    float* x1f    = ws.get<float>(C * NPIX);
    float* x2     = ws.get<float>(C * NPIX);
    float* pooled = ws.get<float>(64);
    float* svec   = ws.get<float>(64);
    float* vbuf   = ws.get<float>(NB * NN);
    float* el     = ws.get<float>(NB * NN * 5);
    float* er     = ws.get<float>(NB * NN * 5);
    float* mbuf   = ws.get<float>(NB * NN * 5);
    float* dbuf   = ws.get<float>(NB * NN * 5);
    float* g3tmp  = ws.get<float>(NB * NN * 192);
    float* hbar   = ws.get<float>(C * NPIX);
    float* uf     = ws.get<float>(C * NPIX);
    // f16 working buffers
    _Float16* th    = ws.get<_Float16>(C * NPIX);
    _Float16* x1s   = ws.get<_Float16>(C * NPIX);
    _Float16* uh    = ws.get<_Float16>(C * NPIX);
    _Float16* feat0 = ws.get<_Float16>(NB * NN * 64);
    _Float16* feat1 = ws.get<_Float16>(NB * NN * 384);
    _Float16* feat2 = ws.get<_Float16>(NB * NN * 640);
    _Float16* zbuf  = ws.get<_Float16>(NB * NN * 640);
    _Float16* zTb   = ws.get<_Float16>((size_t)NB * 5 * 128 * NNP);

    // --- stage weights / input ---
    k_f32_to_f16<<<g1(C * NPIX), 256, 0, stream>>>(F(IN_X), x0h, C * NPIX);
    const int widx[6] = {IN_C1_W1, IN_C1_W2, IN_C12_W1, IN_C12_W2, IN_C22_W1, IN_C22_W2};
    for (int i = 0; i < 6; ++i)
        k_f32_to_f16<<<g1(36864), 256, 0, stream>>>(F(widx[i]), wc[i], 36864);
    k_wT<<<g1(384 * 64), 256, 0, stream>>>(F(IN_G1_W), gwT1, 64, 384);
    k_wT<<<g1(640 * 384), 256, 0, stream>>>(F(IN_G2_W), gwT2, 384, 640);
    k_wT<<<g1(192 * 640), 256, 0, stream>>>(F(IN_G3_W), gwT3, 640, 192);

    dim3 convGrid((NPIX + 63) / 64, 4);  // 183 x 4
    auto dualconv = [&](const _Float16* in16, const _Float16* w1h, int ib1, int ig1, int ibe1,
                        const _Float16* w2h, int ib2, int ig2, int ibe2,
                        float* out32, _Float16* out16) {
        k_conv3x3<<<convGrid, 32, 0, stream>>>(in16, w1h, F(ib1), tconv);
        k_bn_stats<<<64, 256, 0, stream>>>(tconv, F(ig1), F(ibe1), scale, shift);
        k_bn_relu<<<g1(C * NPIX), 256, 0, stream>>>(tconv, scale, shift, nullptr, th);
        k_conv3x3<<<convGrid, 32, 0, stream>>>(th, w2h, F(ib2), tconv);
        k_bn_stats<<<64, 256, 0, stream>>>(tconv, F(ig2), F(ibe2), scale, shift);
        k_bn_relu<<<g1(C * NPIX), 256, 0, stream>>>(tconv, scale, shift, out32, out16);
    };

    // x = dualconv(x0)
    dualconv(x0h, wc[0], IN_C1_B1, IN_C1_G1, IN_C1_BE1,
             wc[1], IN_C1_B2, IN_C1_G2, IN_C1_BE2, xf, th);

    // channel-attention branch
    k_am<<<g1(NPIX), 256, 0, stream>>>(xf, F(IN_AM11_W), x1);
    k_pool<<<64, 256, 0, stream>>>(x1, pooled);
    k_mlp<<<1, 64, 0, stream>>>(pooled, F(IN_MLP_W1), F(IN_MLP_B1), F(IN_MLP_W2), F(IN_MLP_B2), svec);
    k_scale_f16<<<g1(C * NPIX), 256, 0, stream>>>(x1, svec, x1s);
    dualconv(x1s, wc[2], IN_C12_B1, IN_C12_G1, IN_C12_BE1,
             wc[3], IN_C12_B2, IN_C12_G2, IN_C12_BE2, x1f, th);

    // graph branch
    k_am<<<g1(NPIX), 256, 0, stream>>>(xf, F(IN_AM21_W), x2);
    k_blockify<<<g1(NB * NN), 256, 0, stream>>>(x2, feat0, vbuf);

    auto gat = [&](const _Float16* feat, int Fin, const _Float16* gwT,
                   int ial, int iar, int ib, int H, int Fo,
                   _Float16* o16, float* o32, int act) {
        k_gemm_f16<<<dim3(NN / 16, (H * Fo) / 64, NB), 32, 0, stream>>>(feat, gwT, zbuf, NN, H * Fo, Fin);
        k_zT<<<g1(NB * H * Fo * NNP), 256, 0, stream>>>(zbuf, zTb, H, Fo);
        k_eler<<<g1(NB * NN * H), 256, 0, stream>>>(zbuf, F(ial), F(iar), el, er, H, Fo);
        k_attn_stats<<<g1(NB * NN * H), 256, 0, stream>>>(el, er, vbuf, mbuf, dbuf, H);
        k_attn_out<<<dim3(NN / 16, Fo / 64, NB * H), 32, 0, stream>>>(
            zTb, el, er, vbuf, mbuf, dbuf, F(ib), o16, o32, H, Fo, act);
    };
    gat(feat0, 64,  gwT1, IN_G1_AL, IN_G1_AR, IN_G1_B, 3, 128, feat1, nullptr, 1);
    gat(feat1, 384, gwT2, IN_G2_AL, IN_G2_AR, IN_G2_B, 5, 128, feat2, nullptr, 1);
    gat(feat2, 640, gwT3, IN_G3_AL, IN_G3_AR, IN_G3_B, 3, 64, nullptr, g3tmp, 0);
    k_head_mean<<<g1(NB * NN * 64), 256, 0, stream>>>(g3tmp, hbar);
    k_uperm<<<g1(C * NPIX), 256, 0, stream>>>(hbar, uh);
    dualconv(uh, wc[4], IN_C22_B1, IN_C22_G1, IN_C22_BE1,
             wc[5], IN_C22_B2, IN_C22_G2, IN_C22_BE2, uf, th);

    // fuse
    k_final<<<g1(C * NPIX), 256, 0, stream>>>(x1f, uf, F(IN_AM_END_W), (float*)d_out);
}
